// AppearanceLoss_87608742904605
// MI455X (gfx1250) — matrix operand order repaired and checked by reading, not compiled, not run on hardware
//
#include <hip/hip_runtime.h>
#include <hip/hip_bf16.h>

typedef __attribute__((ext_vector_type(16))) _Float16 v16h;
typedef __attribute__((ext_vector_type(8)))  _Float16 v8h;
typedef __attribute__((ext_vector_type(8)))  float    v8f;

#define PATCH    33
#define NPOSMAX  527          // max conv1 positions staged per pass (17 rows * 31)
#define NFEAT    128
#define NPATCH   2048         // 1024 ground + 1024 satellite
#define FEATELEM (1024 * NFEAT)

static __device__ __forceinline__ v8f wmma16(v16h a, v16h b, v8f c) {
  // v_wmma_f32_16x16x32_f16 : D = A(16x32 f16) * B(32x16 f16) + C(16x16 f32)
  return __builtin_amdgcn_wmma_f32_16x16x32_f16(false, a, false, b, (short)0, c,
                                                false, false);
}

// K index held by A-matrix element (vgpr j, half h, lane-group g); contiguous
// order matching B: lanes 0-15 hold K 0..15, lanes 16-31 hold K 16..31.
static __device__ __forceinline__ int aK(int j, int h, int g) {
  return 16 * g + 2 * j + h;
}

__global__ __launch_bounds__(128)
void cnn_patch_features(const float* __restrict__ img_g, const float* __restrict__ img_s,
                        const float* __restrict__ kp_g,  const float* __restrict__ kp_s,
                        const float* __restrict__ w1,    const float* __restrict__ b1,
                        const float* __restrict__ w2,    const float* __restrict__ b2,
                        const float* __restrict__ wl,    const float* __restrict__ bl,
                        float* __restrict__ feat)
{
  __shared__ _Float16 s_patch[3 * PATCH * PATCH];                 //  6534 B
  __shared__ __align__(64) _Float16 s_h1[NPOSMAX * 32];           // 33728 B, [pos][ic]
  __shared__ float    s_gap[64];                                  //   256 B

  const int tid  = threadIdx.x;
  const int lane = tid & 31;
  const int wave = tid >> 5;       // 4 waves
  const int g    = lane >> 4;      // lane group (0/1)
  const int ln   = lane & 15;      // N index inside a 16-wide tile

  // patch id -> (image set, keypoint n, batch image b); m = n*4 + b per set.
  const int m   = blockIdx.x;
  const int set = m >> 10;
  const int mp  = m & 1023;
  const int n   = mp >> 2;
  const int b   = mp & 3;

  const float* img = set ? img_s : img_g;
  const float* kp  = set ? kp_s  : kp_g;

  // crop start: clip(floor(kp*256) - 16, 0, 223); kp = (x, y)
  int sx = (int)floorf(kp[2 * n + 0] * 256.0f) - 16;
  int sy = (int)floorf(kp[2 * n + 1] * 256.0f) - 16;
  sx = sx < 0 ? 0 : (sx > 223 ? 223 : sx);
  sy = sy < 0 ? 0 : (sy > 223 ? 223 : sy);

  // ---- conv1 weights resident in A-layout (2 oc-tiles, K=27 padded to 32) ----
  v16h a1[2];
  #pragma unroll
  for (int t = 0; t < 2; ++t) {
    const int oc = t * 16 + ln;
    #pragma unroll
    for (int j = 0; j < 8; ++j)
      #pragma unroll
      for (int h = 0; h < 2; ++h) {
        const int k = aK(j, h, g);             // k = ic*9 + kh*3 + kw
        a1[t][2 * j + h] = (k < 27) ? (_Float16)w1[oc * 27 + k] : (_Float16)0.0f;
      }
  }
  // ---- conv2 weights resident: this wave owns oc group [wave*16, wave*16+16) ----
  v16h a2[9];
  {
    const int oc = wave * 16 + ln;
    #pragma unroll
    for (int c = 0; c < 9; ++c)                // c = kh*3 + kw
      #pragma unroll
      for (int j = 0; j < 8; ++j)
        #pragma unroll
        for (int h = 0; h < 2; ++h) {
          const int ic = aK(j, h, g);          // K inside chunk = input channel
          a2[c][2 * j + h] = (_Float16)w2[oc * 288 + ic * 9 + c];
        }
  }
  // ---- bias fragments resident (C/D layout: vgpr r -> M = r + 8*lanegroup) ----
  float bias1[2][8], bias2[8];
  #pragma unroll
  for (int r = 0; r < 8; ++r) {
    bias1[0][r] = b1[r + 8 * g];
    bias1[1][r] = b1[16 + r + 8 * g];
    bias2[r]    = b2[wave * 16 + r + 8 * g];
  }

  // ---- stage patch into LDS as f16 ----
  for (int idx = tid; idx < 3 * PATCH * PATCH; idx += 128) {
    const int c  = idx / (PATCH * PATCH);
    const int r  = idx - c * (PATCH * PATCH);
    const int py = r / PATCH;
    const int px = r - py * PATCH;
    s_patch[idx] =
        (_Float16)img[((b * 3 + c) << 16) + ((sy + py) << 8) + (sx + px)];
  }
  if (tid < 64) s_gap[tid] = 0.0f;
  __syncthreads();

  float vsum[8];
  #pragma unroll
  for (int r = 0; r < 8; ++r) vsum[r] = 0.0f;

  // Two spatial passes so the [pos][ic] staging buffer fits in LDS:
  //  pass0: conv1 rows 0..16  -> conv2 out rows 0..7  (120 positions)
  //  pass1: conv1 rows 16..30 -> conv2 out rows 8..14 (105 positions)
  #pragma unroll
  for (int pass = 0; pass < 2; ++pass) {
    const int r0     = pass ? 16 : 0;
    const int nrow1  = pass ? 15 : 17;
    const int npos1  = nrow1 * 31;
    const int ntile1 = (npos1 + 15) >> 4;

    // ---------- conv1: [32 x 27] @ im2col[27 x npos1]; B built once/tile ----
    for (int pt = wave; pt < ntile1; pt += 4) {
      const int p  = (pt << 4) + ln;
      const int pc = (p < npos1) ? p : (npos1 - 1);     // clamp, mask on store
      const int ly = pc / 31;
      const int lx = pc - ly * 31;
      const int base = (r0 + ly) * PATCH + lx;          // + ic*1089 + kh*33 + kw
      v16h bm;
      #pragma unroll
      for (int e = 0; e < 16; ++e) {
        // B element e holds K = 16*g + e; clamp K>=27 to 26 (A is zero there,
        // so the garbage operand contributes 0). Offsets fold to constants.
        const int k0   = e;                              // g == 0
        const int k1   = (16 + e > 26) ? 26 : (16 + e);  // g == 1, clamped
        const int off0 = (k0 / 9) * 1089 + ((k0 % 9) / 3) * PATCH + (k0 % 3);
        const int off1 = (k1 / 9) * 1089 + ((k1 % 9) / 3) * PATCH + (k1 % 3);
        bm[e] = s_patch[base + (g ? off1 : off0)];
      }
      #pragma unroll
      for (int oc2 = 0; oc2 < 2; ++oc2) {
        v8f acc = {};
        acc = wmma16(a1[oc2], bm, acc);
        v8h hv;
        #pragma unroll
        for (int r = 0; r < 8; ++r) {
          float x = acc[r] + bias1[oc2][r];
          x = x > 0.0f ? x : 0.0f;
          hv[r] = (_Float16)x;
        }
        // lane covers 8 consecutive oc -> one 16B store, h1 layout [pos][ic]
        if (p < npos1)
          *(v8h*)&s_h1[p * 32 + (oc2 << 4) + 8 * g] = hv;
      }
    }
    __syncthreads();

    // ---------- conv2: [64 x 288] @ im2col[288 x npos2], stride 2 ----------
    const int noy    = pass ? 7 : 8;
    const int npos2  = noy * 15;
    const int ntile2 = (npos2 + 15) >> 4;
    for (int t = 0; t < ntile2; ++t) {
      const int p   = (t << 4) + ln;
      const int pc  = (p < npos2) ? p : (npos2 - 1);
      const int oyl = pc / 15;
      const int ox  = pc - oyl * 15;
      // Rolling load -> WMMA with alternating accumulators: only one B fragment
      // live at a time (a2[] keeps 72 VGPRs resident), chain depth halved.
      v8f acc0 = {}, acc1 = {};
      #pragma unroll
      for (int c = 0; c < 9; ++c) {
        const int kh   = c / 3;
        const int kw   = c - kh * 3;
        const int spos = (2 * oyl + kh) * 31 + 2 * ox + kw;
        // B fragment = 32 contiguous bytes: element e -> ic = 16*g + e
        const v16h bm = *(const v16h*)&s_h1[spos * 32 + 16 * g];
        if (c & 1) acc1 = wmma16(a2[c], bm, acc1);
        else       acc0 = wmma16(a2[c], bm, acc0);
      }
      #pragma unroll
      for (int r = 0; r < 8; ++r) {
        float x = acc0[r] + acc1[r] + bias2[r];
        x = x > 0.0f ? x : 0.0f;
        vsum[r] += (p < npos2) ? x : 0.0f;
      }
    }
    __syncthreads();
  }

  // ---- global average pool: register partials -> LDS (ds_add_f32) ----
  #pragma unroll
  for (int r = 0; r < 8; ++r)
    atomicAdd(&s_gap[wave * 16 + r + 8 * g], vsum[r]);
  __syncthreads();

  // ---- linear 64 -> 128 (one output per thread, vectorized weight reads) ----
  {
    const float4* wlv = (const float4*)(wl + tid * 64);
    float acc = bl[tid];
    const float inv = 1.0f / 225.0f;
    #pragma unroll
    for (int c4 = 0; c4 < 16; ++c4) {
      const float4 w4 = wlv[c4];
      acc += w4.x * (s_gap[4 * c4 + 0] * inv);
      acc += w4.y * (s_gap[4 * c4 + 1] * inv);
      acc += w4.z * (s_gap[4 * c4 + 2] * inv);
      acc += w4.w * (s_gap[4 * c4 + 3] * inv);
    }
    feat[m * NFEAT + tid] = acc;
  }
}

__global__ void zero_out_kernel(float* out) { out[0] = 0.0f; }

__global__ __launch_bounds__(256)
void mse_kernel(const float* __restrict__ feat, float* __restrict__ out)
{
  __shared__ float red[256];
  const float4* fa = (const float4*)feat;
  const float4* fb = (const float4*)(feat + FEATELEM);
  float s = 0.0f;
  for (int i = blockIdx.x * blockDim.x + threadIdx.x; i < FEATELEM / 4;
       i += gridDim.x * blockDim.x) {
    const float4 a = fa[i];
    const float4 bq = fb[i];
    const float d0 = a.x - bq.x, d1 = a.y - bq.y;
    const float d2 = a.z - bq.z, d3 = a.w - bq.w;
    s += d0 * d0 + d1 * d1 + d2 * d2 + d3 * d3;
  }
  red[threadIdx.x] = s;
  __syncthreads();
  for (int st = 128; st > 0; st >>= 1) {
    if (threadIdx.x < st) red[threadIdx.x] += red[threadIdx.x + st];
    __syncthreads();
  }
  if (threadIdx.x == 0)
    atomicAdd(out, red[0] * (1.0f / (float)FEATELEM));
}

extern "C" void kernel_launch(void* const* d_in, const int* in_sizes, int n_in,
                              void* d_out, int out_size, void* d_ws, size_t ws_size,
                              hipStream_t stream) {
  const float* img_g = (const float*)d_in[0];
  const float* img_s = (const float*)d_in[1];
  const float* kp_g  = (const float*)d_in[2];
  const float* kp_s  = (const float*)d_in[3];
  const float* w1    = (const float*)d_in[4];
  const float* b1    = (const float*)d_in[5];
  const float* w2    = (const float*)d_in[6];
  const float* b2    = (const float*)d_in[7];
  const float* wl    = (const float*)d_in[8];
  const float* bl    = (const float*)d_in[9];
  float* out  = (float*)d_out;
  float* feat = (float*)d_ws;   // needs NPATCH*NFEAT*4 = 1 MB scratch

  cnn_patch_features<<<NPATCH, 128, 0, stream>>>(img_g, img_s, kp_g, kp_s,
                                                 w1, b1, w2, b2, wl, bl, feat);
  zero_out_kernel<<<1, 1, 0, stream>>>(out);
  mse_kernel<<<64, 256, 0, stream>>>(feat, out);
}
